// SelfAttention_v2_26431228739824
// MI455X (gfx1250) — matrix operand order repaired and checked by reading, not compiled, not run on hardware
//
#include <hip/hip_runtime.h>

#define SEQ  8192
#define DIN  768
#define DOUT 64

typedef __attribute__((ext_vector_type(16))) __bf16 v16bf;
typedef __attribute__((ext_vector_type(8)))  __bf16 bf8v;
typedef __attribute__((ext_vector_type(8)))  float  v8f;
typedef __attribute__((ext_vector_type(4)))  float  f32x4;

// softmax scale folded into Wq at conversion time: 1/sqrt(64) * log2(e),
// so phase 2 runs a pure base-2 softmax (v_exp_f32, no extra multiplies).
#define QSCALE (0.125f * 1.44269504088896340736f)

static __device__ __forceinline__ v16bf cat8(bf8v a, bf8v b) {
  v16bf r;
#pragma unroll
  for (int i = 0; i < 8; ++i) { r[i] = a[i]; r[i + 8] = b[i]; }
  return r;
}

// ---------------------------------------------------------------------------
// Phase 0: bulk fp32 -> bf16 conversion with optional scale (hardware v_cvt).
// Each thread converts 8 elements: 2x b128 load -> 1x b128 store.
// ---------------------------------------------------------------------------
__global__ __launch_bounds__(256)
void cvt_bf16_kernel(const float* __restrict__ src,
                     __bf16* __restrict__ dst, int n8, float scale) {
  const int i = blockIdx.x * blockDim.x + threadIdx.x;
  if (i >= n8) return;
  const f32x4* s = (const f32x4*)src + (size_t)i * 2;
  const f32x4 a = s[0];
  const f32x4 b = s[1];
  bf8v r;
#pragma unroll
  for (int j = 0; j < 4; ++j) {
    r[j]     = (__bf16)(a[j] * scale);
    r[j + 4] = (__bf16)(b[j] * scale);
  }
  ((bf8v*)dst)[i] = r;
}

// ---------------------------------------------------------------------------
// Phase 1: Q = x @ (s*Wq)^T, K = x @ Wk^T, V = x @ Wv^T  (all-bf16 GEMM)
// Q, K stored row-major [SEQ, 64]; V stored transposed [64, SEQ].
// grid = (SEQ/16, 3), block = 128 (4 waves); wave w -> output cols [16w,16w+16)
// Loop body is pure b128-load + WMMA.
// ---------------------------------------------------------------------------
__global__ __launch_bounds__(128)
void qkv_proj_kernel(const __bf16* __restrict__ xb,
                     const __bf16* __restrict__ Wb,   // [3][64][768] bf16
                     __bf16* __restrict__ Qb,
                     __bf16* __restrict__ Kb,
                     __bf16* __restrict__ Vt) {
  const int wave = threadIdx.x >> 5;
  const int lane = threadIdx.x & 31;
  const int l15  = lane & 15;
  const int hi   = lane >> 4;          // 0: lanes 0-15, 1: lanes 16-31
  const int t    = blockIdx.x;         // 16-row tile of x
  const int my   = blockIdx.y;         // 0=Q, 1=K, 2=V
  const __bf16* W = Wb + (size_t)my * DOUT * DIN;
  const int col  = wave * 16 + l15;    // output column (B lane / D lane)
  const int arow = t * 16 + l15;       // x row (A lane)

  const bf8v* aptr = (const bf8v*)(xb + (size_t)arow * DIN + (hi ? 16 : 0));
  const bf8v* bptr = (const bf8v*)(W + (size_t)col * DIN + (hi ? 16 : 0));

  v8f acc = {0.f, 0.f, 0.f, 0.f, 0.f, 0.f, 0.f, 0.f};
#pragma unroll 4
  for (int kc = 0; kc < DIN / 32; ++kc) {
    // A tile (16x32 of x): lane holds 16 contiguous dims off..off+15
    const v16bf a = cat8(aptr[0], aptr[1]);
    // B tile (32x16 of W^T): lane holds 16 contiguous k for column `col`
    const v16bf b = cat8(bptr[0], bptr[1]);
    aptr += 4;  // advance 32 dims
    bptr += 4;
    acc = __builtin_amdgcn_wmma_f32_16x16x32_bf16(false, a, false, b,
                                                  (short)0, acc, false, false);
  }

  // D layout: lane holds column `col`, rows mbase+0..7
  const int mbase = t * 16 + (hi ? 8 : 0);
  if (my == 2) {
    // V^T [64, SEQ]: 8 consecutive seq positions -> one 16B store
    bf8v pk;
#pragma unroll
    for (int v = 0; v < 8; ++v) pk[v] = (__bf16)acc[v];
    *(bf8v*)(Vt + (size_t)col * SEQ + mbase) = pk;
  } else {
    __bf16* dst = (my == 0) ? Qb : Kb;
#pragma unroll
    for (int v = 0; v < 8; ++v)
      dst[(size_t)(mbase + v) * DOUT + col] = (__bf16)acc[v];
  }
}

// ---------------------------------------------------------------------------
// Phase 2: flash attention, fully transposed formulation, base-2 softmax.
// One wave owns 16 query rows; loops over SEQ keys in tiles of 32.
//   S^T = K_tile @ Q^T   (D layout: lane = one query, 8 keys)
//   online softmax: local reduce + shfl_xor(16); probabilities via exp2f
//   O^T += V^T @ P^T     (P^T rebuilt into B layout via 2 shfl_xor(16))
// grid = SEQ/16/4, block = 128 (4 waves, 1 query-tile each)
// ---------------------------------------------------------------------------
__global__ __launch_bounds__(128)
void flash_attn_kernel(const __bf16* __restrict__ Qb,
                       const __bf16* __restrict__ Kb,
                       const __bf16* __restrict__ Vt,
                       float* __restrict__ out) {
  const int wave  = threadIdx.x >> 5;
  const int lane  = threadIdx.x & 31;
  const int l15   = lane & 15;
  const int hi    = lane >> 4;
  const int qt    = blockIdx.x * 4 + wave;
  const int qbase = qt * 16;

  // B = Q^T (64x16), two 32-dim chunks; lane = query column, 16 contiguous dims
  v16bf bq[2];
#pragma unroll
  for (int c = 0; c < 2; ++c) {
    const bf8v* p =
        (const bf8v*)(Qb + (size_t)(qbase + l15) * DOUT + c * 32 + (hi ? 16 : 0));
    bq[c] = cat8(p[0], p[1]);
  }

  v8f oacc[4];
#pragma unroll
  for (int d = 0; d < 4; ++d)
    oacc[d] = (v8f){0.f, 0.f, 0.f, 0.f, 0.f, 0.f, 0.f, 0.f};
  float mrun = -3.0e38f;   // running max (base-2 exponent domain)
  float lrun = 0.0f;

#pragma unroll 1
  for (int kb = 0; kb < SEQ; kb += 32) {
    // ---- S^T tiles: keys [kb,kb+16) and [kb+16,kb+32) x 16 queries ----
    v8f st[2];
#pragma unroll
    for (int sub = 0; sub < 2; ++sub) {
      v8f c = {0.f, 0.f, 0.f, 0.f, 0.f, 0.f, 0.f, 0.f};
#pragma unroll
      for (int ck = 0; ck < 2; ++ck) {
        const bf8v* p = (const bf8v*)(Kb +
            (size_t)(kb + sub * 16 + l15) * DOUT + ck * 32 + (hi ? 16 : 0));
        const v16bf a = cat8(p[0], p[1]);  // dims {off..+15}
        c = __builtin_amdgcn_wmma_f32_16x16x32_bf16(false, a, false, bq[ck],
                                                    (short)0, c, false, false);
      }
      st[sub] = c;  // softmax scale pre-folded into Q
    }

    // ---- online softmax (per query; lane pair l and l+16 stay in sync) ----
    float tmax = st[0][0];
#pragma unroll
    for (int i = 1; i < 8; ++i) tmax = fmaxf(tmax, st[0][i]);
#pragma unroll
    for (int i = 0; i < 8; ++i) tmax = fmaxf(tmax, st[1][i]);
    tmax = fmaxf(tmax, __shfl_xor(tmax, 16, 32));
    const float mnew  = fmaxf(mrun, tmax);
    const float alpha = exp2f(mrun - mnew);
    v8f e0, e1;
    float ssum = 0.f;
#pragma unroll
    for (int i = 0; i < 8; ++i) { e0[i] = exp2f(st[0][i] - mnew); ssum += e0[i]; }
#pragma unroll
    for (int i = 0; i < 8; ++i) { e1[i] = exp2f(st[1][i] - mnew); ssum += e1[i]; }
    ssum += __shfl_xor(ssum, 16, 32);
    lrun = lrun * alpha + ssum;
    mrun = mnew;

    // ---- rebuild P^T (32 keys x 16 queries) into WMMA B layout ----
    // lane<16 needs keys 0..15: own ST0 + partner ST0
    // lane>=16 needs keys 16..31: partner ST1 + own ST1
    v16bf pB;
#pragma unroll
    for (int i = 0; i < 8; ++i) {
      const float o0 = __shfl_xor(e0[i], 16, 32);
      const float o1 = __shfl_xor(e1[i], 16, 32);
      pB[i]     = (__bf16)(hi ? o1 : e0[i]);
      pB[i + 8] = (__bf16)(hi ? e1[i] : o0);
    }

    // ---- O^T += V^T @ P^T, 4 head-dim chunks of 16 ----
#pragma unroll
    for (int dc = 0; dc < 4; ++dc) {
      const bf8v* p = (const bf8v*)(Vt +
          (size_t)(dc * 16 + l15) * SEQ + kb + (hi ? 16 : 0));
      const v16bf a = cat8(p[0], p[1]);  // keys {off..+15}
      v8f cacc = oacc[dc] * alpha;       // rescale running accumulator
      oacc[dc] = __builtin_amdgcn_wmma_f32_16x16x32_bf16(false, a, false, pB,
                                                         (short)0, cacc, false, false);
    }

    if (kb + 32 < SEQ) {
      __builtin_prefetch(Kb + (size_t)(kb + 32 + l15) * DOUT, 0, 0);
      __builtin_prefetch(Vt + (size_t)l15 * SEQ + kb + 32, 0, 0);
    }
  }

  // ---- finalize: out[q, d] = O^T[d, q] / l ----
  const float inv = 1.0f / lrun;
  float* dst = out + (size_t)(qbase + l15) * DOUT;
#pragma unroll
  for (int dc = 0; dc < 4; ++dc) {
    f32x4 r0, r1;
#pragma unroll
    for (int i = 0; i < 4; ++i) {
      r0[i] = oacc[dc][i] * inv;
      r1[i] = oacc[dc][i + 4] * inv;
    }
    f32x4* o = (f32x4*)(dst + dc * 16 + (hi ? 8 : 0));
    o[0] = r0;
    o[1] = r1;
  }
}

extern "C" void kernel_launch(void* const* d_in, const int* in_sizes, int n_in,
                              void* d_out, int out_size, void* d_ws, size_t ws_size,
                              hipStream_t stream) {
  const float* x  = (const float*)d_in[0];
  const float* Wq = (const float*)d_in[1];
  const float* Wk = (const float*)d_in[2];
  const float* Wv = (const float*)d_in[3];

  // workspace layout (bf16):
  //   xb [SEQ,768]   : 12.0 MiB
  //   Wb [3][64,768] : 288 KiB
  //   Qb [SEQ,64], Kb [SEQ,64], Vt [64,SEQ] : 1 MiB each
  __bf16* xb = (__bf16*)d_ws;
  __bf16* Wb = xb + (size_t)SEQ * DIN;
  __bf16* Qb = Wb + (size_t)3 * DOUT * DIN;
  __bf16* Kb = Qb + (size_t)SEQ * DOUT;
  __bf16* Vt = Kb + (size_t)SEQ * DOUT;
  float* out = (float*)d_out;

  const int nx8 = SEQ * DIN / 8;        // x elements / 8
  const int nw8 = DOUT * DIN / 8;       // per weight matrix
  cvt_bf16_kernel<<<(nx8 + 255) / 256, 256, 0, stream>>>(x, xb, nx8, 1.0f);
  cvt_bf16_kernel<<<(nw8 + 255) / 256, 256, 0, stream>>>(Wq, Wb, nw8, QSCALE);
  cvt_bf16_kernel<<<(nw8 + 255) / 256, 256, 0, stream>>>(Wk, Wb + (size_t)DOUT * DIN, nw8, 1.0f);
  cvt_bf16_kernel<<<(nw8 + 255) / 256, 256, 0, stream>>>(Wv, Wb + (size_t)2 * DOUT * DIN, nw8, 1.0f);

  qkv_proj_kernel<<<dim3(SEQ / 16, 3), 128, 0, stream>>>(xb, Wb, Qb, Kb, Vt);
  flash_attn_kernel<<<dim3(SEQ / 16 / 4), 128, 0, stream>>>(Qb, Kb, Vt, out);
}
